// LSTM_3848290697395
// MI455X (gfx1250) — matrix operand order repaired and checked by reading, not compiled, not run on hardware
//
#include <hip/hip_runtime.h>
#include <math.h>

typedef __bf16 bf16_t;
typedef __attribute__((ext_vector_type(16))) __bf16 v16bf;
typedef __attribute__((ext_vector_type(8)))  __bf16 v8bf;
typedef __attribute__((ext_vector_type(8)))  float  v8f;

#define B_   512
#define T_   96
#define D_   64
#define H_   256
#define O_   24
#define G4H  1024
#define KTH  (T_ * H_)   // 24576

#if defined(__HIP_DEVICE_COMPILE__) && __has_builtin(__builtin_amdgcn_global_load_async_to_lds_b32)
#define HAVE_ASYNC_LDS 1
#else
#define HAVE_ASYNC_LDS 0
#endif

typedef __attribute__((address_space(1))) int as1_int;
typedef __attribute__((address_space(3))) int as3_int;

static __device__ __forceinline__ float fast_tanh(float x) {
#if defined(__HIP_DEVICE_COMPILE__) && __has_builtin(__builtin_amdgcn_tanhf)
  return __builtin_amdgcn_tanhf(x);
#elif defined(__HIP_DEVICE_COMPILE__) && __has_builtin(__builtin_amdgcn_tanh_f32)
  return __builtin_amdgcn_tanh_f32(x);
#else
  return tanhf(x);
#endif
}

static __device__ __forceinline__ float sigmoid_f(float x) {
  // sigmoid(x) = 0.5*tanh(x/2) + 0.5  -> single V_TANH_F32 + FMA
  return __builtin_fmaf(0.5f, fast_tanh(0.5f * x), 0.5f);
}

static __device__ __forceinline__ void async_copy_u32(const unsigned* src, unsigned* lds_dst) {
#if HAVE_ASYNC_LDS
  __builtin_amdgcn_global_load_async_to_lds_b32(
      (as1_int*)src, (as3_int*)lds_dst, 0, 0);
#else
  *lds_dst = *src;
#endif
}

static __device__ __forceinline__ void async_fence_lds() {
#if HAVE_ASYNC_LDS
  asm volatile("s_wait_asynccnt 0x0" ::: "memory");
#endif
}

// ---------------- small prep kernels ----------------

__global__ void cvt_f32_bf16_k(const float* __restrict__ in, bf16_t* __restrict__ out, int n) {
  int i = blockIdx.x * blockDim.x + threadIdx.x;
  if (i < n) out[i] = (bf16_t)in[i];
}

__global__ void bias_sum_k(const float* __restrict__ a, const float* __restrict__ b,
                           float* __restrict__ o, int n) {
  int i = blockIdx.x * blockDim.x + threadIdx.x;
  if (i < n) o[i] = a[i] + b[i];
}

__global__ void init_state_k(const float* __restrict__ h0, const float* __restrict__ c0,
                             bf16_t* __restrict__ hstate, float* __restrict__ cstate) {
  int i = blockIdx.x * blockDim.x + threadIdx.x;   // over 2*B*H
  const int BH = B_ * H_;
  if (i >= 2 * BH) return;
  int l = i / BH;
  int r = i - l * BH;
  hstate[(size_t)(l * 2 + 0) * BH + r] = (bf16_t)h0[i];   // parity 0 holds h0
  cstate[i] = c0[i];
}

// ---------------- LSTM step (WMMA bf16) ----------------

struct StepTask {
  const bf16_t* zx;      // x-part of A matrix, rows [B], K1 cols, row stride ldzx
  int ldzx;
  int K1;
  const bf16_t* hin;     // recurrent h input [B,256]
  const bf16_t* Wx;      // [1024,K1] row-major
  const bf16_t* Wh;      // [1024,256] row-major
  const float*  bsum;    // [1024] b_ih + b_hh
  float*        cstate;  // [B,256]
  bf16_t*       hout;    // [B,256]
  bf16_t*       outslab; // optional [B][T][H] slab at time t (may be null)
  int           outStride;
};

static __device__ __forceinline__ void lstm_step_body(const StepTask& tk) {
  const int PITCH = 260;                 // uints per LDS row = 1040 B (16B aligned)
  __shared__ unsigned zs[16 * PITCH];    // up to 16 x 512 bf16

  const int m0   = blockIdx.x * 16;
  const int lane = threadIdx.x & 31;
  const int wave = threadIdx.x >> 5;
  const int K1   = tk.K1;
  const int Ktot = K1 + H_;

  // Stage z = [x_part | h_part] rows m0..m0+15 into LDS (async-to-LDS when available)
  {
    const int Ku = Ktot >> 1;
    for (int idx = threadIdx.x; idx < 16 * Ku; idx += blockDim.x) {
      int r  = idx / Ku;
      int ku = idx - r * Ku;
      int k  = ku << 1;
      int row = m0 + r;
      const bf16_t* src = (k < K1) ? (tk.zx + (size_t)row * tk.ldzx + k)
                                   : (tk.hin + (size_t)row * H_ + (k - K1));
      async_copy_u32((const unsigned*)src, &zs[r * PITCH + ku]);
    }
  }
  async_fence_lds();
  __syncthreads();

  const int n0   = (blockIdx.y * 8 + wave) * 16;  // hidden-column tile (per gate)
  const int ncol = lane & 15;
  const int hi   = lane >> 4;

  v8f acc[4];
  {
    float bi = tk.bsum[0 * H_ + n0 + ncol];
    float bf = tk.bsum[1 * H_ + n0 + ncol];
    float bg = tk.bsum[2 * H_ + n0 + ncol];
    float bo = tk.bsum[3 * H_ + n0 + ncol];
    acc[0] = (v8f){bi, bi, bi, bi, bi, bi, bi, bi};
    acc[1] = (v8f){bf, bf, bf, bf, bf, bf, bf, bf};
    acc[2] = (v8f){bg, bg, bg, bg, bg, bg, bg, bg};
    acc[3] = (v8f){bo, bo, bo, bo, bo, bo, bo, bo};
  }

  const bf16_t* zrow = (const bf16_t*)(zs + (lane & 15) * PITCH);

  auto loadA = [&](int kb) -> v16bf {
    v8bf a0 = *(const v8bf*)(zrow + kb + (hi << 4));
    v8bf a1 = *(const v8bf*)(zrow + kb + 8 + (hi << 4));
    return __builtin_shufflevector(a0, a1, 0, 1, 2, 3, 4, 5, 6, 7,
                                           8, 9, 10, 11, 12, 13, 14, 15);
  };
  auto loadB = [&](int g, int kb) -> v16bf {
    int kk = kb + (hi << 4);
    int ng = g * H_ + n0 + ncol;
    const bf16_t* wp = (kk < K1) ? (tk.Wx + (size_t)ng * K1 + kk)
                                 : (tk.Wh + (size_t)ng * H_ + (kk - K1));
    return *(const v16bf*)wp;
  };

  // Software-pipelined K loop: next iteration's fragments load during this
  // iteration's 4 WMMAs so s_wait covers a whole prefetch clause.
  v16bf a  = loadA(0);
  v16bf b0 = loadB(0, 0), b1 = loadB(1, 0), b2 = loadB(2, 0), b3 = loadB(3, 0);
  for (int kb = 0; kb < Ktot; kb += 32) {
    const int kn = (kb + 32 < Ktot) ? (kb + 32) : kb;
    v16bf an  = loadA(kn);
    v16bf c0n = loadB(0, kn), c1n = loadB(1, kn), c2n = loadB(2, kn), c3n = loadB(3, kn);
    acc[0] = __builtin_amdgcn_wmma_f32_16x16x32_bf16(false, a, false, b0, (short)0, acc[0], false, false);
    acc[1] = __builtin_amdgcn_wmma_f32_16x16x32_bf16(false, a, false, b1, (short)0, acc[1], false, false);
    acc[2] = __builtin_amdgcn_wmma_f32_16x16x32_bf16(false, a, false, b2, (short)0, acc[2], false, false);
    acc[3] = __builtin_amdgcn_wmma_f32_16x16x32_bf16(false, a, false, b3, (short)0, acc[3], false, false);
    a = an; b0 = c0n; b1 = c1n; b2 = c2n; b3 = c3n;
  }

  // Pointwise LSTM update; lane holds col n0+ncol, rows m0 + hi*8 + r
  const int mb  = hi * 8;
  const int col = n0 + ncol;
#pragma unroll
  for (int r = 0; r < 8; ++r) {
    int row = m0 + mb + r;
    size_t idx = (size_t)row * H_ + col;
    float iv = sigmoid_f(acc[0][r]);
    float fv = sigmoid_f(acc[1][r]);
    float gv = fast_tanh(acc[2][r]);
    float ov = sigmoid_f(acc[3][r]);
    float cn = fv * tk.cstate[idx] + iv * gv;
    tk.cstate[idx] = cn;
    float hv = ov * fast_tanh(cn);
    bf16_t hb = (bf16_t)hv;
    tk.hout[idx] = hb;
    if (tk.outslab) tk.outslab[(size_t)row * tk.outStride + col] = hb;
  }
}

__global__ __launch_bounds__(256) void lstm_step1_k(StepTask t0) {
  lstm_step_body(t0);
}

// Fused: layer0 step t+1 (z=0) and layer1 step t (z=1) are independent.
__global__ __launch_bounds__(256) void lstm_step2_k(StepTask t0, StepTask t1) {
  if (blockIdx.z == 0) lstm_step_body(t0);
  else                 lstm_step_body(t1);
}

// ---------------- fc1: [512,24576]bf16 @ [24576,256] -> tanh -> [512,256] f32 ----------------

__global__ __launch_bounds__(256) void fc1_k(const bf16_t* __restrict__ A,
                                             const bf16_t* __restrict__ W,
                                             const float* __restrict__ bias,
                                             float* __restrict__ hid) {
  const int CH    = 1024;               // K chunk (bf16 elems)
  const int PITCH = 516;                // uints per LDS row = 2064 B (16B aligned)
  __shared__ unsigned zs[16 * PITCH];   // 33 KB

  const int m0   = blockIdx.x * 16;
  const int lane = threadIdx.x & 31;
  const int wave = threadIdx.x >> 5;
  const int n0   = (blockIdx.y * 8 + wave) * 16;
  const int ncol = lane & 15;
  const int hi   = lane >> 4;

  v8f acc = (v8f){0, 0, 0, 0, 0, 0, 0, 0};
  const bf16_t* zrow = (const bf16_t*)(zs + (lane & 15) * PITCH);
  const bf16_t* wrow = W + (size_t)(n0 + ncol) * KTH;

  for (int kbase = 0; kbase < KTH; kbase += CH) {
    __syncthreads();
    for (int idx = threadIdx.x; idx < 16 * (CH / 2); idx += blockDim.x) {
      int r  = idx / (CH / 2);
      int ku = idx - r * (CH / 2);
      async_copy_u32((const unsigned*)(A + (size_t)(m0 + r) * KTH + kbase + (ku << 1)),
                     &zs[r * PITCH + ku]);
    }
    async_fence_lds();
    __syncthreads();

    v16bf b = *(const v16bf*)(wrow + kbase + (hi << 4));
#pragma unroll 4
    for (int kb = 0; kb < CH; kb += 32) {
      const int kn = (kb + 32 < CH) ? (kb + 32) : kb;
      v16bf bn = *(const v16bf*)(wrow + kbase + kn + (hi << 4));
      v8bf a0 = *(const v8bf*)(zrow + kb + (hi << 4));
      v8bf a1 = *(const v8bf*)(zrow + kb + 8 + (hi << 4));
      v16bf a = __builtin_shufflevector(a0, a1, 0, 1, 2, 3, 4, 5, 6, 7,
                                                8, 9, 10, 11, 12, 13, 14, 15);
      acc = __builtin_amdgcn_wmma_f32_16x16x32_bf16(
          false, a, false, b, (short)0, acc, false, false);
      b = bn;
    }
  }

  const int mb = hi * 8, col = n0 + ncol;
  float bv = bias[col];
#pragma unroll
  for (int r = 0; r < 8; ++r) {
    hid[(size_t)(m0 + mb + r) * 256 + col] = fast_tanh(acc[r] + bv);
  }
}

// ---------------- fc2: [512,256] @ [256,24] -> relu ----------------

__global__ void fc2_k(const float* __restrict__ hid, const float* __restrict__ W,
                      const float* __restrict__ bias, float* __restrict__ out) {
  int i = blockIdx.x * blockDim.x + threadIdx.x;  // b*O + o
  if (i >= B_ * O_) return;
  int b = i / O_, o = i - b * O_;
  const float* hrow = hid + (size_t)b * 256;
  const float* wr   = W + (size_t)o * 256;
  float s = bias[o];
#pragma unroll 8
  for (int k = 0; k < 256; ++k) s += hrow[k] * wr[k];
  out[i] = fmaxf(s, 0.0f);
}

// ---------------- launch ----------------

extern "C" void kernel_launch(void* const* d_in, const int* in_sizes, int n_in,
                              void* d_out, int out_size, void* d_ws, size_t ws_size,
                              hipStream_t stream) {
  (void)in_sizes; (void)n_in; (void)out_size; (void)ws_size;

  const float* x     = (const float*)d_in[0];
  const float* h0    = (const float*)d_in[1];
  const float* c0    = (const float*)d_in[2];
  const float* W_ih0 = (const float*)d_in[3];
  const float* W_hh0 = (const float*)d_in[4];
  const float* b_ih0 = (const float*)d_in[5];
  const float* b_hh0 = (const float*)d_in[6];
  const float* W_ih1 = (const float*)d_in[7];
  const float* W_hh1 = (const float*)d_in[8];
  const float* b_ih1 = (const float*)d_in[9];
  const float* b_hh1 = (const float*)d_in[10];
  const float* fc1_w = (const float*)d_in[11];
  const float* fc1_b = (const float*)d_in[12];
  const float* fc2_w = (const float*)d_in[13];
  const float* fc2_b = (const float*)d_in[14];

  char* p = (char*)d_ws;
  auto take = [&](size_t bytes) -> char* {
    char* r = p;
    p += (bytes + 255) & ~(size_t)255;
    return r;
  };

  bf16_t* xbf    = (bf16_t*)take((size_t)B_ * T_ * D_ * 2);
  bf16_t* wih0b  = (bf16_t*)take((size_t)G4H * D_ * 2);
  bf16_t* whh0b  = (bf16_t*)take((size_t)G4H * H_ * 2);
  bf16_t* wih1b  = (bf16_t*)take((size_t)G4H * H_ * 2);
  bf16_t* whh1b  = (bf16_t*)take((size_t)G4H * H_ * 2);
  bf16_t* fc1wb  = (bf16_t*)take((size_t)256 * KTH * 2);
  float*  bsum0  = (float*)take((size_t)G4H * 4);
  float*  bsum1  = (float*)take((size_t)G4H * 4);
  bf16_t* hstate = (bf16_t*)take((size_t)2 * 2 * B_ * H_ * 2);  // [layer][parity][B*H]
  float*  cstate = (float*)take((size_t)2 * B_ * H_ * 4);
  bf16_t* out1   = (bf16_t*)take((size_t)B_ * T_ * H_ * 2);
  float*  hid    = (float*)take((size_t)B_ * 256 * 4);

  auto CV = [&](const float* src, bf16_t* dst, int n) {
    cvt_f32_bf16_k<<<(n + 255) / 256, 256, 0, stream>>>(src, dst, n);
  };
  CV(x, xbf, B_ * T_ * D_);
  CV(W_ih0, wih0b, G4H * D_);
  CV(W_hh0, whh0b, G4H * H_);
  CV(W_ih1, wih1b, G4H * H_);
  CV(W_hh1, whh1b, G4H * H_);
  CV(fc1_w, fc1wb, 256 * KTH);
  bias_sum_k<<<4, 256, 0, stream>>>(b_ih0, b_hh0, bsum0, G4H);
  bias_sum_k<<<4, 256, 0, stream>>>(b_ih1, b_hh1, bsum1, G4H);
  init_state_k<<<(2 * B_ * H_ + 255) / 256, 256, 0, stream>>>(h0, c0, hstate, cstate);

  auto hst = [&](int layer, int par) -> bf16_t* {
    return hstate + ((size_t)layer * 2 + par) * B_ * H_;
  };
  auto mkL0 = [&](int t) -> StepTask {
    StepTask s;
    s.zx = xbf + (size_t)t * D_; s.ldzx = T_ * D_; s.K1 = D_;
    s.hin = hst(0, t & 1);
    s.Wx = wih0b; s.Wh = whh0b; s.bsum = bsum0;
    s.cstate = cstate;
    s.hout = hst(0, 1 - (t & 1));
    s.outslab = nullptr; s.outStride = 0;
    return s;
  };
  auto mkL1 = [&](int t) -> StepTask {
    StepTask s;
    s.zx = hst(0, 1 - (t & 1)); s.ldzx = H_; s.K1 = H_;   // layer0 output at step t
    s.hin = hst(1, t & 1);
    s.Wx = wih1b; s.Wh = whh1b; s.bsum = bsum1;
    s.cstate = cstate + (size_t)B_ * H_;
    s.hout = hst(1, 1 - (t & 1));
    s.outslab = out1 + (size_t)t * H_; s.outStride = T_ * H_;
    return s;
  };

  const dim3 sblk(256);
  lstm_step1_k<<<dim3(32, 2, 1), sblk, 0, stream>>>(mkL0(0));
  for (int t = 0; t < T_ - 1; ++t) {
    // layer0 step t+1 and layer1 step t are independent -> fused launch
    lstm_step2_k<<<dim3(32, 2, 2), sblk, 0, stream>>>(mkL0(t + 1), mkL1(t));
  }
  lstm_step1_k<<<dim3(32, 2, 1), sblk, 0, stream>>>(mkL1(T_ - 1));

  fc1_k<<<dim3(32, 2), 256, 0, stream>>>(out1, fc1wb, fc1_b, hid);
  fc2_k<<<(B_ * O_ + 255) / 256, 256, 0, stream>>>(hid, fc2_w, fc2_b, (float*)d_out);
}
